// DecoderBlock_37623913513565
// MI455X (gfx1250) — matrix operand order repaired and checked by reading, not compiled, not run on hardware
//
#include <hip/hip_runtime.h>

// ---------------- problem constants ----------------
#define B_ 4
#define S_ 2048
#define D_ 768
#define H_ 12
#define DH_ 64
#define DFF_ 3072
#define NTOK_ (B_ * S_)   // 8192

typedef __attribute__((ext_vector_type(16))) __bf16 v16bf;
typedef __attribute__((ext_vector_type(8)))  float  v8f;

__device__ inline v8f zero8() {
  v8f z = {0.f, 0.f, 0.f, 0.f, 0.f, 0.f, 0.f, 0.f};
  return z;
}

__device__ inline v8f wmma_bf16(v16bf a, v16bf b, v8f c) {
  // D = A(16x32 bf16) * B(32x16 bf16) + C(16x16 f32)
  return __builtin_amdgcn_wmma_f32_16x16x32_bf16(
      false, a, false, b, (short)0, c, false, false);
}

// fp32 -> bf16 round-to-nearest-even
__device__ inline unsigned short f2bf(float f) {
  union { float f; unsigned int u; } c;
  c.f = f;
  unsigned int u = c.u;
  u += 0x7FFFu + ((u >> 16) & 1u);
  return (unsigned short)(u >> 16);
}

// LDS byte offset of a __shared__ object (generic addr low 32 bits)
__device__ inline unsigned ldsoff(const void* p) {
  return (unsigned)(unsigned long long)p;
}

// CDNA5 async copy: 16 bytes per lane, global -> LDS, tracked by ASYNCcnt
__device__ inline void async_b128(unsigned lds, const unsigned short* g) {
  asm volatile("global_load_async_to_lds_b128 %0, %1, off"
               :: "v"(lds), "v"(g) : "memory");
}
__device__ inline void wait_async0() {
  asm volatile("s_wait_asynccnt 0" ::: "memory");
}
__device__ inline void wait_async4() {
  asm volatile("s_wait_asynccnt 4" ::: "memory");
}

// A-fragment (16x32 bf16): lane 0-15 = row M=lane, K {koff..+7, koff+16..+23};
// lane 16-31 = same rows, K {+8..15, +24..31}.  LDS row-major, stride mult of 8.
__device__ inline v16bf load_frag_A(const unsigned short* lds, int stride,
                                    int m_row, int k_off, int lane) {
  const int l15 = lane & 15;
  const int hi  = lane >> 4;
  const unsigned short* p = lds + (m_row + l15) * stride + k_off + hi * 8;
  union { v16bf v; uint4 q[2]; } r;
  r.q[0] = *(const uint4*)(p);
  r.q[1] = *(const uint4*)(p + 16);
  return r.v;
}

// B-fragment (32x16 bf16): lane 0-15 = col N=lane, K {0..7, 16..23}; lane 16-31 = same
// cols, K {8..15, 24..31}.  LDS holds B transposed: [N][K] row-major, stride mult of 8.
__device__ inline v16bf load_frag_B(const unsigned short* lds, int stride,
                                    int n_col, int k_off, int lane) {
  const int l15 = lane & 15;
  const int hi  = lane >> 4;
  const unsigned short* p = lds + (n_col + l15) * stride + k_off + hi * 8;
  union { v16bf v; uint4 q[2]; } r;
  r.q[0] = *(const uint4*)(p);
  r.q[1] = *(const uint4*)(p + 16);
  return r.v;
}

// =====================================================================
// One-time per-launch precision prep
// =====================================================================
__global__ __launch_bounds__(256) void f32_to_bf16_kernel(
    const float* __restrict__ in, unsigned short* __restrict__ out, int n4) {
  const int i = (blockIdx.x * 256 + threadIdx.x);
  if (i < n4) {
    float4 v = *(const float4*)(in + (size_t)i * 4);
    unsigned short* o = out + (size_t)i * 4;
    o[0] = f2bf(v.x); o[1] = f2bf(v.y); o[2] = f2bf(v.z); o[3] = f2bf(v.w);
  }
}

// W[K][N] fp32 -> WT[N][K] bf16 (32x32 tiles via LDS)
__global__ __launch_bounds__(256) void transpose_bf16_kernel(
    const float* __restrict__ in, unsigned short* __restrict__ out,
    int K, int N) {
  __shared__ float tile[32][33];
  const int k0 = blockIdx.y * 32, n0 = blockIdx.x * 32;
  const int tx = threadIdx.x & 31, ty = threadIdx.x >> 5;
#pragma unroll
  for (int i = 0; i < 4; ++i) {
    const int k = ty + i * 8;
    tile[k][tx] = in[(size_t)(k0 + k) * N + n0 + tx];
  }
  __syncthreads();
#pragma unroll
  for (int i = 0; i < 4; ++i) {
    const int n = ty + i * 8;
    out[(size_t)(n0 + n) * K + k0 + tx] = f2bf(tile[tx][n]);
  }
}

// =====================================================================
// GEMM: C[M,N] = act(A[M,K] @ BT[N,K]^T + bias).  A, BT bf16; C fp32 or
// bf16.  128x128 tile, 8 wave32 waves (4x2), 32x64 per wave, K-chunk 32,
// double-buffered async global->LDS staging.
// =====================================================================
__global__ __launch_bounds__(256) void gemm_kernel(
    const unsigned short* __restrict__ A, const unsigned short* __restrict__ BT,
    const float* __restrict__ bias, void* __restrict__ C,
    int M, int N, int K, int relu, int outBf16) {
  __shared__ __align__(16) unsigned short As[2][128 * 40];
  __shared__ __align__(16) unsigned short Bs[2][128 * 40];

  const int t    = threadIdx.x;
  const int lane = t & 31;
  const int wv   = t >> 5;
  const int wm   = wv >> 1;      // 0..3
  const int wn   = wv & 1;       // 0..1
  const int m0   = blockIdx.y * 128;
  const int n0   = blockIdx.x * 128;
  const int l15  = lane & 15;
  const int hi   = lane >> 4;

  v8f acc[2][4];
#pragma unroll
  for (int i = 0; i < 2; ++i)
#pragma unroll
    for (int j = 0; j < 4; ++j) acc[i][j] = zero8();

  // 4 async b128 issues per thread per chunk (2 for A, 2 for B)
  auto issue = [&](int buf, int k0) {
#pragma unroll
    for (int i = 0; i < 2; ++i) {
      const int c   = i * 256 + t;       // 512 chunks of 8 shorts
      const int row = c >> 2;
      const int kk  = (c & 3) * 8;
      async_b128(ldsoff(&As[buf][row * 40 + kk]),
                 A + (size_t)(m0 + row) * K + k0 + kk);
    }
#pragma unroll
    for (int i = 0; i < 2; ++i) {
      const int c   = i * 256 + t;
      const int row = c >> 2;
      const int kk  = (c & 3) * 8;
      async_b128(ldsoff(&Bs[buf][row * 40 + kk]),
                 BT + (size_t)(n0 + row) * K + k0 + kk);
    }
  };

  int cb = 0;
  issue(0, 0);
  for (int k0 = 0; k0 < K; k0 += 32) {
    const bool has = (k0 + 32) < K;
    if (has) issue(cb ^ 1, k0 + 32);
    if (has) wait_async4(); else wait_async0();
    __syncthreads();

    v16bf a0 = load_frag_A(As[cb], 40, wm * 32 + 0, 0, lane);
    v16bf a1 = load_frag_A(As[cb], 40, wm * 32 + 16, 0, lane);
#pragma unroll
    for (int j = 0; j < 4; ++j) {
      v16bf bf = load_frag_B(Bs[cb], 40, wn * 64 + j * 16, 0, lane);
      acc[0][j] = wmma_bf16(a0, bf, acc[0][j]);
      acc[1][j] = wmma_bf16(a1, bf, acc[1][j]);
    }
    __syncthreads();
    cb ^= 1;
  }

  // epilogue: lane = column n, VGPR r = row (hi half adds 8)
  float* Cf = (float*)C;
  unsigned short* Cb = (unsigned short*)C;
#pragma unroll
  for (int j = 0; j < 4; ++j) {
    const int n  = n0 + wn * 64 + j * 16 + l15;
    const float bv = bias ? bias[n] : 0.0f;
#pragma unroll
    for (int i = 0; i < 2; ++i) {
#pragma unroll
      for (int r = 0; r < 8; ++r) {
        const int m = m0 + wm * 32 + i * 16 + hi * 8 + r;
        float v = acc[i][j][r] + bv;
        if (relu) v = fmaxf(v, 0.0f);
        if (outBf16) Cb[(size_t)m * N + n] = f2bf(v);
        else         Cf[(size_t)m * N + n] = v;
      }
    }
  }
}

// =====================================================================
// Attention pass 1: per-(b,h) softmax stats over the QUERY axis.
// wT[k,q] = KV[k,:] . Q[q,:] / 8 ; m[k] = max_q, l[k] = sum_q exp(w-m).
// Block owns 128 k-rows (wave w owns 16); q streamed in 128-chunks with
// double-buffered async staging.  Q, KV are bf16.
// =====================================================================
__global__ __launch_bounds__(256) void attn_stats_kernel(
    const unsigned short* __restrict__ Q, const unsigned short* __restrict__ KV,
    float* __restrict__ mOut, float* __restrict__ lOut, int causal) {
  __shared__ __align__(16) unsigned short kvs[128 * 72];
  __shared__ __align__(16) unsigned short qs[2][128 * 72];

  const int t    = threadIdx.x;
  const int lane = t & 31;
  const int wv   = t >> 5;
  const int l15  = lane & 15;
  const int hi   = lane >> 4;
  const int kblk = blockIdx.x * 128;
  const int bh   = blockIdx.y;
  const int b    = bh / H_;
  const int h    = bh % H_;
  const unsigned short* Qp  = Q  + (size_t)b * S_ * D_ + h * DH_;
  const unsigned short* KVp = KV + (size_t)b * S_ * D_ + h * DH_;

  // stage KV block rows kblk..kblk+128 (64 bf16 each) via async copies
#pragma unroll
  for (int i = 0; i < 4; ++i) {
    const int c   = i * 256 + t;    // 1024 chunks of 8 shorts
    const int row = c >> 3;
    const int kk  = (c & 7) * 8;
    async_b128(ldsoff(&kvs[row * 72 + kk]),
               KVp + (size_t)(kblk + row) * D_ + kk);
  }
  wait_async0();
  __syncthreads();
  const v16bf aKV0 = load_frag_A(kvs, 72, wv * 16, 0, lane);
  const v16bf aKV1 = load_frag_A(kvs, 72, wv * 16, 32, lane);

  auto issueQ = [&](int buf, int qc) {
#pragma unroll
    for (int i = 0; i < 4; ++i) {
      const int c   = i * 256 + t;
      const int row = c >> 3;
      const int kk  = (c & 7) * 8;
      async_b128(ldsoff(&qs[buf][row * 72 + kk]),
                 Qp + (size_t)(qc + row) * D_ + kk);
    }
  };

  const float scale = 0.125f;          // 1/sqrt(64)
  const float LOG2E = 1.44269504088896f;
  float m_run[8], l_run[8];
#pragma unroll
  for (int r = 0; r < 8; ++r) { m_run[r] = -3.0e38f; l_run[r] = 0.0f; }

  const int q0 = causal ? kblk : 0;
  int cb = 0;
  issueQ(0, q0);
  for (int qc = q0; qc < S_; qc += 128) {
    const bool has = (qc + 128) < S_;
    if (has) issueQ(cb ^ 1, qc + 128);
    if (has) wait_async4(); else wait_async0();
    __syncthreads();

    float m_ln[8], l_ln[8];
#pragma unroll
    for (int r = 0; r < 8; ++r) { m_ln[r] = -3.0e38f; l_ln[r] = 0.0f; }

#pragma unroll
    for (int j = 0; j < 8; ++j) {
      v16bf b0 = load_frag_B(qs[cb], 72, j * 16, 0, lane);
      v16bf b1 = load_frag_B(qs[cb], 72, j * 16, 32, lane);
      v8f w = zero8();
      w = wmma_bf16(aKV0, b0, w);
      w = wmma_bf16(aKV1, b1, w);
      const int qcol = qc + j * 16 + l15;
#pragma unroll
      for (int r = 0; r < 8; ++r) {
        const int krow = kblk + wv * 16 + hi * 8 + r;
        const bool masked = causal && (krow > qcol);
        if (!masked) {
          const float v  = w[r] * scale;
          const float mn = fmaxf(m_ln[r], v);
          l_ln[r] = l_ln[r] * exp2f((m_ln[r] - mn) * LOG2E) +
                    exp2f((v - mn) * LOG2E);
          m_ln[r] = mn;
        }
      }
    }
    // merge (m,l) across the 16 lanes that hold the same k-rows
#pragma unroll
    for (int off = 1; off < 16; off <<= 1) {
#pragma unroll
      for (int r = 0; r < 8; ++r) {
        const float mo = __shfl_xor(m_ln[r], off, 32);
        const float lo = __shfl_xor(l_ln[r], off, 32);
        const float mn = fmaxf(m_ln[r], mo);
        l_ln[r] = l_ln[r] * exp2f((m_ln[r] - mn) * LOG2E) +
                  lo * exp2f((mo - mn) * LOG2E);
        m_ln[r] = mn;
      }
    }
    // fold into running stats
#pragma unroll
    for (int r = 0; r < 8; ++r) {
      const float mn = fmaxf(m_run[r], m_ln[r]);
      l_run[r] = l_run[r] * exp2f((m_run[r] - mn) * LOG2E) +
                 l_ln[r] * exp2f((m_ln[r] - mn) * LOG2E);
      m_run[r] = mn;
    }
    __syncthreads();
    cb ^= 1;
  }

  if (l15 == 0) {
    const size_t base = (size_t)bh * S_ + kblk + wv * 16 + hi * 8;
#pragma unroll
    for (int r = 0; r < 8; ++r) {
      mOut[base + r] = m_run[r];
      lOut[base + r] = l_run[r];
    }
  }
}

// =====================================================================
// Attention pass 2: O[q,:] = sum_k exp(w[q,k]-m[k])/l[k] * KV[k,:].
// Block owns 128 q rows (wave w owns 16); k streamed in 32-chunks.
// Q, KV bf16; O written fp32 merged into (B,S,D) at column h*64.
// =====================================================================
__global__ __launch_bounds__(256) void attn_out_kernel(
    const unsigned short* __restrict__ Q, const unsigned short* __restrict__ KV,
    const float* __restrict__ mIn, const float* __restrict__ lIn,
    float* __restrict__ O, int causal) {
  __shared__ __align__(16) unsigned short qs[128 * 72];
  __shared__ __align__(16) unsigned short kv_rm[32 * 72];   // [k][dh]
  __shared__ __align__(16) unsigned short kv_tr[64 * 40];   // [dh][k]
  __shared__ __align__(16) unsigned short ps[8 * 16 * 40];  // per-wave P tile

  const int t    = threadIdx.x;
  const int lane = t & 31;
  const int wv   = t >> 5;
  const int l15  = lane & 15;
  const int hi   = lane >> 4;
  const int qblk = blockIdx.x * 128;
  const int bh   = blockIdx.y;
  const int b    = bh / H_;
  const int h    = bh % H_;
  const unsigned short* Qp  = Q  + (size_t)b * S_ * D_ + h * DH_;
  const unsigned short* KVp = KV + (size_t)b * S_ * D_ + h * DH_;

  // stage Q block rows qblk..qblk+128 via async copies
#pragma unroll
  for (int i = 0; i < 4; ++i) {
    const int c   = i * 256 + t;
    const int row = c >> 3;
    const int kk  = (c & 7) * 8;
    async_b128(ldsoff(&qs[row * 72 + kk]),
               Qp + (size_t)(qblk + row) * D_ + kk);
  }
  wait_async0();
  __syncthreads();
  const v16bf aQ0 = load_frag_A(qs, 72, wv * 16, 0, lane);
  const v16bf aQ1 = load_frag_A(qs, 72, wv * 16, 32, lane);

  v8f accO[4];
#pragma unroll
  for (int j = 0; j < 4; ++j) accO[j] = zero8();

  const float scale = 0.125f;
  const float LOG2E = 1.44269504088896f;
  unsigned short* pW = ps + wv * 16 * 40;
  const int kend = causal ? (qblk + 128) : S_;

  for (int kc = 0; kc < kend; kc += 32) {
    __syncthreads();
    {  // stage KV chunk rows kc..kc+32: row-major copy + transposed scatter
      const int row = t >> 3;          // 0..31
      const int db  = (t & 7) * 8;     // 0..56
      uint4 v = *(const uint4*)(KVp + (size_t)(kc + row) * D_ + db);
      *(uint4*)(kv_rm + row * 72 + db) = v;
      const unsigned int dw[4] = {v.x, v.y, v.z, v.w};
#pragma unroll
      for (int i = 0; i < 4; ++i) {
        kv_tr[(db + 2 * i + 0) * 40 + row] = (unsigned short)(dw[i] & 0xFFFFu);
        kv_tr[(db + 2 * i + 1) * 40 + row] = (unsigned short)(dw[i] >> 16);
      }
    }
    __syncthreads();

    // GEMM1: w[16q x 32k] = Q(16x64) @ KV^T(64x32); then P = exp/l
#pragma unroll
    for (int j = 0; j < 2; ++j) {
      v16bf b0 = load_frag_B(kv_rm, 72, j * 16, 0, lane);
      v16bf b1 = load_frag_B(kv_rm, 72, j * 16, 32, lane);
      v8f w = zero8();
      w = wmma_bf16(aQ0, b0, w);
      w = wmma_bf16(aQ1, b1, w);
      const int kcol  = kc + j * 16 + l15;
      const float mk  = mIn[(size_t)bh * S_ + kcol];
      const float ilk = 1.0f / lIn[(size_t)bh * S_ + kcol];
#pragma unroll
      for (int r = 0; r < 8; ++r) {
        const int qrow = qblk + wv * 16 + hi * 8 + r;
        float p = 0.0f;
        if (!(causal && (kcol > qrow))) {
          p = exp2f((w[r] * scale - mk) * LOG2E) * ilk;
        }
        pW[(hi * 8 + r) * 40 + j * 16 + l15] = f2bf(p);
      }
    }
    __syncthreads();

    // GEMM2: O(16x64) += P(16x32) @ KV(32x64)
    const v16bf aP = load_frag_A(pW, 40, 0, 0, lane);
#pragma unroll
    for (int j = 0; j < 4; ++j) {
      v16bf bK = load_frag_B(kv_tr, 40, j * 16, 0, lane);
      accO[j] = wmma_bf16(aP, bK, accO[j]);
    }
  }

  // store O merged into (B,S,D) at column h*64
#pragma unroll
  for (int j = 0; j < 4; ++j) {
    const int dh = j * 16 + l15;
#pragma unroll
    for (int r = 0; r < 8; ++r) {
      const int q = qblk + wv * 16 + hi * 8 + r;
      O[(size_t)(b * S_ + q) * D_ + h * DH_ + dh] = accO[j][r];
    }
  }
}

// =====================================================================
// Row-wise residual + LayerNorm: out = LN(A + R) * gamma + beta.
// Emits fp32 (next residual) and optionally bf16 (next GEMM operand).
// =====================================================================
__global__ __launch_bounds__(256) void ln_residual_kernel(
    const float* __restrict__ A, const float* __restrict__ R,
    const float* __restrict__ gamma, const float* __restrict__ beta,
    float* __restrict__ out, unsigned short* __restrict__ outb) {
  const int row  = blockIdx.x;
  const int t    = threadIdx.x;
  const int lane = t & 31;
  const int wv   = t >> 5;
  const float* a = A + (size_t)row * D_;
  const float* r = R + (size_t)row * D_;

  float v[3];
  float s1 = 0.0f, s2 = 0.0f;
#pragma unroll
  for (int i = 0; i < 3; ++i) {
    const int c = t + i * 256;
    v[i] = a[c] + r[c];
    s1 += v[i];
    s2 += v[i] * v[i];
  }
#pragma unroll
  for (int off = 1; off < 32; off <<= 1) {
    s1 += __shfl_xor(s1, off, 32);
    s2 += __shfl_xor(s2, off, 32);
  }
  __shared__ float red1[8], red2[8];
  if (lane == 0) { red1[wv] = s1; red2[wv] = s2; }
  __syncthreads();
  float t1 = 0.0f, t2 = 0.0f;
#pragma unroll
  for (int w = 0; w < 8; ++w) { t1 += red1[w]; t2 += red2[w]; }
  const float mu   = t1 * (1.0f / D_);
  const float var  = t2 * (1.0f / D_) - mu * mu;
  const float rstd = rsqrtf(var + 1e-5f);
#pragma unroll
  for (int i = 0; i < 3; ++i) {
    const int c = t + i * 256;
    const float y = (v[i] - mu) * rstd * gamma[c] + beta[c];
    out[(size_t)row * D_ + c] = y;
    if (outb) outb[(size_t)row * D_ + c] = f2bf(y);
  }
}

// =====================================================================
extern "C" void kernel_launch(void* const* d_in, const int* in_sizes, int n_in,
                              void* d_out, int out_size, void* d_ws, size_t ws_size,
                              hipStream_t stream) {
  const float* x   = (const float*)d_in[0];
  // d_in[1] = mask (causal handled analytically)
  const float* Wq1 = (const float*)d_in[2];
  const float* Wv1 = (const float*)d_in[3];
  const float* g1  = (const float*)d_in[4];
  const float* be1 = (const float*)d_in[5];
  const float* Wq2 = (const float*)d_in[6];
  const float* Wv2 = (const float*)d_in[7];
  const float* g2  = (const float*)d_in[8];
  const float* be2 = (const float*)d_in[9];
  const float* Wf1 = (const float*)d_in[10];
  const float* bf1 = (const float*)d_in[11];
  const float* Wf2 = (const float*)d_in[12];
  const float* bf2 = (const float*)d_in[13];
  const float* gf  = (const float*)d_in[14];
  const float* bef = (const float*)d_in[15];
  float* out = (float*)d_out;

  // ---- workspace carve-up (deterministic bump allocator) ----
  char* wp = (char*)d_ws;
  auto alloc = [&](size_t bytes) {
    char* p = wp;
    wp += (bytes + 255) & ~(size_t)255;
    return p;
  };
  const size_t SD  = (size_t)NTOK_ * D_;
  const size_t SF  = (size_t)NTOK_ * DFF_;
  unsigned short* xb    = (unsigned short*)alloc(SD * 2);
  unsigned short* Qb    = (unsigned short*)alloc(SD * 2);
  unsigned short* KVb   = (unsigned short*)alloc(SD * 2);
  unsigned short* x1b   = (unsigned short*)alloc(SD * 2);
  unsigned short* x2b   = (unsigned short*)alloc(SD * 2);
  unsigned short* Hb    = (unsigned short*)alloc(SF * 2);
  float*          x1f   = (float*)alloc(SD * 4);
  float*          x2f   = (float*)alloc(SD * 4);
  float*          Of    = (float*)alloc(SD * 4);
  unsigned short* Wq1T  = (unsigned short*)alloc((size_t)D_ * D_ * 2);
  unsigned short* Wv1T  = (unsigned short*)alloc((size_t)D_ * D_ * 2);
  unsigned short* Wq2T  = (unsigned short*)alloc((size_t)D_ * D_ * 2);
  unsigned short* Wv2T  = (unsigned short*)alloc((size_t)D_ * D_ * 2);
  unsigned short* Wf1T  = (unsigned short*)alloc((size_t)D_ * DFF_ * 2);
  unsigned short* Wf2T  = (unsigned short*)alloc((size_t)DFF_ * D_ * 2);
  float*          mBuf  = (float*)alloc((size_t)B_ * H_ * S_ * 4);
  float*          lBuf  = (float*)alloc((size_t)B_ * H_ * S_ * 4);

  const dim3 blk(256);
  const dim3 gProj(D_ / 128, NTOK_ / 128);    // (6, 64)
  const dim3 gFf1(DFF_ / 128, NTOK_ / 128);   // (24, 64)
  const dim3 gAttn(S_ / 128, B_ * H_);        // (16, 48)

  // ---- one-time precision prep ----
  f32_to_bf16_kernel<<<(SD / 4 + 255) / 256, blk, 0, stream>>>(x, xb, (int)(SD / 4));
  transpose_bf16_kernel<<<dim3(D_ / 32, D_ / 32), blk, 0, stream>>>(Wq1, Wq1T, D_, D_);
  transpose_bf16_kernel<<<dim3(D_ / 32, D_ / 32), blk, 0, stream>>>(Wv1, Wv1T, D_, D_);
  transpose_bf16_kernel<<<dim3(D_ / 32, D_ / 32), blk, 0, stream>>>(Wq2, Wq2T, D_, D_);
  transpose_bf16_kernel<<<dim3(D_ / 32, D_ / 32), blk, 0, stream>>>(Wv2, Wv2T, D_, D_);
  transpose_bf16_kernel<<<dim3(DFF_ / 32, D_ / 32), blk, 0, stream>>>(Wf1, Wf1T, D_, DFF_);
  transpose_bf16_kernel<<<dim3(D_ / 32, DFF_ / 32), blk, 0, stream>>>(Wf2, Wf2T, DFF_, D_);

  // ---- attention block 1 (causal) ----
  gemm_kernel<<<gProj, blk, 0, stream>>>(xb, Wq1T, nullptr, Qb,  NTOK_, D_, D_, 0, 1);
  gemm_kernel<<<gProj, blk, 0, stream>>>(xb, Wv1T, nullptr, KVb, NTOK_, D_, D_, 0, 1);
  attn_stats_kernel<<<gAttn, blk, 0, stream>>>(Qb, KVb, mBuf, lBuf, 1);
  attn_out_kernel<<<gAttn, blk, 0, stream>>>(Qb, KVb, mBuf, lBuf, Of, 1);
  ln_residual_kernel<<<NTOK_, blk, 0, stream>>>(Of, x, g1, be1, x1f, x1b);

  // ---- attention block 2 (no mask) ----
  gemm_kernel<<<gProj, blk, 0, stream>>>(x1b, Wq2T, nullptr, Qb,  NTOK_, D_, D_, 0, 1);
  gemm_kernel<<<gProj, blk, 0, stream>>>(x1b, Wv2T, nullptr, KVb, NTOK_, D_, D_, 0, 1);
  attn_stats_kernel<<<gAttn, blk, 0, stream>>>(Qb, KVb, mBuf, lBuf, 0);
  attn_out_kernel<<<gAttn, blk, 0, stream>>>(Qb, KVb, mBuf, lBuf, Of, 0);
  ln_residual_kernel<<<NTOK_, blk, 0, stream>>>(Of, x1f, g2, be2, x2f, x2b);

  // ---- feed-forward ----
  gemm_kernel<<<gFf1, blk, 0, stream>>>(x2b, Wf1T, bf1, Hb, NTOK_, DFF_, D_, 1, 1);
  gemm_kernel<<<gProj, blk, 0, stream>>>(Hb, Wf2T, bf2, Of, NTOK_, D_, DFF_, 0, 0);
  ln_residual_kernel<<<NTOK_, blk, 0, stream>>>(Of, x2f, gf, bef, out, nullptr);
}
